// GCNPlugin_14534169330304
// MI455X (gfx1250) — compile-verified, matrix operand order checked by reading, not run on hardware
//
#include <hip/hip_runtime.h>
#include <hip/hip_fp16.h>

typedef __attribute__((ext_vector_type(16))) _Float16 v16h;
typedef __attribute__((ext_vector_type(8)))  float    v8f;

#define N_NODES  50000
#define IN_CH    256
#define HID_CH   128
#define OUT_CH   1000
#define OUT_PAD  1024
#define TEXT_DIM 768
#define N_EDGES  800000

// ---------------------------------------------------------------------------
// WMMA layouts (CDNA5 ISA, wave32):
// A (16x32 f16 from row-major [M x K]):
//   lanes 0-15 : row M=lane,    halves 0-7 = K k0+0..7,  8-15 = K k0+16..23
//   lanes 16-31: row M=lane-16, halves 0-7 = K k0+8..15, 8-15 = K k0+24..31
// B (32x16 f16): mirrored, lane selects column. We PRE-PACK B into fragment
// order [ntile][ktile][lane][16 halves] so the GEMM loads one contiguous
// 32-byte chunk per lane (2x global_load_b128) instead of 16 scalar u16s.
// C/D (16x16 f32, 8 regs): elem v -> row = v + 8*(lane>>4), col = lane&15.
// ---------------------------------------------------------------------------
__device__ __forceinline__ v16h load_a16(const _Float16* __restrict__ A, int lda,
                                         int row0, int k0, int lane) {
  const int m  = lane & 15;
  const int kh = (lane >> 4) << 3;                     // 0 or 8
  const _Float16* p = A + (size_t)(row0 + m) * lda + k0 + kh;
  v16h a;
#pragma unroll
  for (int i = 0; i < 8; ++i) { a[i] = p[i]; a[i + 8] = p[i + 16]; }
  return a;
}

__device__ __forceinline__ v16h load_bp(const _Float16* __restrict__ Bp,
                                        int KT, int nt, int kt, int lane) {
  return *(const v16h*)(Bp + ((((size_t)nt * KT + kt) * 32 + lane) << 4));
}

// ---------------- small utility kernels ----------------
__global__ void k_fill1(float* __restrict__ d, int n) {
  int i = blockIdx.x * blockDim.x + threadIdx.x;
  if (i < n) d[i] = 1.0f;                               // self-loop degree
}

__global__ void k_deg_count(const long long* __restrict__ dst, float* __restrict__ deg) {
  int e = blockIdx.x * blockDim.x + threadIdx.x;
  if (e < N_EDGES) unsafeAtomicAdd(&deg[(int)dst[e]], 1.0f);
}

__global__ void k_rsqrt(const float* __restrict__ deg, float* __restrict__ dis, int n) {
  int i = blockIdx.x * blockDim.x + threadIdx.x;
  if (i < n) dis[i] = rsqrtf(deg[i]);                   // deg >= 1 always
}

__global__ void k_f2h(const float* __restrict__ in, _Float16* __restrict__ out, int n) {
  int i = blockIdx.x * blockDim.x + threadIdx.x;
  if (i < n) out[i] = (_Float16)in[i];
}

// Pack row-major f32 B[K x N_src] into WMMA-fragment-ordered f16:
// dst[((nt*KT + kt)*32 + lane)*16 + i], zero-padding columns >= N_src.
__global__ void k_pack_b(const float* __restrict__ src, _Float16* __restrict__ dst,
                         int KT, int NT, int N_src) {
  int idx = blockIdx.x * blockDim.x + threadIdx.x;
  if (idx >= KT * NT * 512) return;
  int i    = idx & 15;
  int lane = (idx >> 4) & 31;
  int rest = idx >> 9;
  int kt   = rest % KT;
  int nt   = rest / KT;
  int n    = nt * 16 + (lane & 15);
  int krel = ((lane >> 4) << 3) + (i & 7) + ((i >> 3) << 4);
  int k    = kt * 32 + krel;
  dst[idx] = (n < N_src) ? (_Float16)src[(size_t)k * N_src + n] : (_Float16)0.0f;
}

// base[j] = bc[j] + sum_k text[k] * Wc[128+k, j]   (text vector is row-broadcast)
__global__ void k_base(const float* __restrict__ text, const float* __restrict__ Wc,
                       const float* __restrict__ bc, float* __restrict__ base) {
  int j = blockIdx.x * blockDim.x + threadIdx.x;
  if (j >= OUT_PAD) return;
  float s = 0.0f;
  if (j < OUT_CH) {
    s = bc[j];
    for (int k = 0; k < TEXT_DIM; ++k)
      s += text[k] * Wc[(size_t)(HID_CH + k) * OUT_CH + j];
  }
  base[j] = s;
}

// ---------------- WMMA GEMMs ----------------
// C[M x 128] = A[M x K] * B[K x 128]; one wave per 16x16 tile, 8 waves per
// block cover the 8 N-tiles of one M-tile. B pre-packed, K compile-time.
template <int K>
__global__ void k_gemm_hid(const _Float16* __restrict__ A, const _Float16* __restrict__ Bp,
                           float* __restrict__ C) {
  constexpr int KT = K / 32;
  const int lane = threadIdx.x & 31;
  const int wave = threadIdx.x >> 5;                    // 0..7 -> N tile
  const int row0 = blockIdx.x * 16;
  v8f acc = {};
#pragma unroll
  for (int kt = 0; kt < KT; ++kt) {
    v16h a = load_a16(A, K, row0, kt * 32, lane);
    v16h b = load_bp(Bp, KT, wave, kt, lane);
    acc = __builtin_amdgcn_wmma_f32_16x16x32_f16(false, a, false, b,
                                                 (short)0, acc, false, false);
  }
  const int n  = lane & 15;
  const int mo = (lane >> 4) << 3;
#pragma unroll
  for (int v = 0; v < 8; ++v)
    C[(size_t)(row0 + v + mo) * HID_CH + wave * 16 + n] = acc[v];
}

// out[M x 1000] = A[M x 128] * Wc_top + base. Each wave owns 4 N-tiles so the
// A fragment is loaded once per 4 WMMAs. 16 wave-slots cover 64 padded tiles.
__global__ void k_classifier(const _Float16* __restrict__ A, const _Float16* __restrict__ Bp,
                             const float* __restrict__ base, float* __restrict__ out) {
  constexpr int KT = HID_CH / 32;                       // 4
  const int lane = threadIdx.x & 31;
  const int wave = threadIdx.x >> 5;
  const int row0 = blockIdx.x * 16;
  const int slot = blockIdx.y * 8 + wave;               // 0..15
  const int nt0  = slot * 4;
  v8f acc[4] = {};
#pragma unroll
  for (int kt = 0; kt < KT; ++kt) {
    v16h a = load_a16(A, HID_CH, row0, kt * 32, lane);
#pragma unroll
    for (int j = 0; j < 4; ++j) {
      v16h b = load_bp(Bp, KT, nt0 + j, kt, lane);
      acc[j] = __builtin_amdgcn_wmma_f32_16x16x32_f16(false, a, false, b,
                                                      (short)0, acc[j], false, false);
    }
  }
  const int n  = lane & 15;
  const int mo = (lane >> 4) << 3;
#pragma unroll
  for (int j = 0; j < 4; ++j) {
    const int col = (nt0 + j) * 16 + n;
    if (col < OUT_CH) {
      const float bb = base[col];
#pragma unroll
      for (int v = 0; v < 8; ++v)
        out[(size_t)(row0 + v + mo) * OUT_CH + col] = acc[j][v] + bb;
    }
  }
}

// ---------------- message passing ----------------
// Self-loop term doubles as the zero-init of the aggregation buffer (graph-
// replay safe): out[i,:] = h[i,:] * dis[i]^2  ( = h[i]/deg[i] ).
__global__ void k_self_init(const float* __restrict__ h, const float* __restrict__ dis,
                            float* __restrict__ out) {
  int idx = blockIdx.x * blockDim.x + threadIdx.x;      // N_NODES*128
  if (idx >= N_NODES * HID_CH) return;
  int i = idx >> 7;
  float w = dis[i];
  out[idx] = h[idx] * w * w;
}

// One wave per edge, each lane scatter-adds 4 channels (128 = 32 lanes * 4).
__global__ void k_edge_agg(const float* __restrict__ h, const long long* __restrict__ src,
                           const long long* __restrict__ dst, const float* __restrict__ dis,
                           float* __restrict__ out) {
  long long t = (long long)blockIdx.x * blockDim.x + threadIdx.x;
  int e = (int)(t >> 5);
  if (e >= N_EDGES) return;
  int lane = (int)(t & 31);
  int s = (int)src[e], d = (int)dst[e];
  float w = dis[s] * dis[d];
  const float4 v = *(const float4*)(h + (size_t)s * HID_CH + lane * 4);
  float* o = out + (size_t)d * HID_CH + lane * 4;
  unsafeAtomicAdd(o + 0, v.x * w);
  unsafeAtomicAdd(o + 1, v.y * w);
  unsafeAtomicAdd(o + 2, v.z * w);
  unsafeAtomicAdd(o + 3, v.w * w);
}

// bias (+ optional ReLU) and convert to f16 for the next WMMA GEMM.
__global__ void k_bias_act(const float* __restrict__ in, const float* __restrict__ bias,
                           _Float16* __restrict__ outh, int relu) {
  int idx = blockIdx.x * blockDim.x + threadIdx.x;
  if (idx >= N_NODES * HID_CH) return;
  float v = in[idx] + bias[idx & (HID_CH - 1)];
  if (relu) v = fmaxf(v, 0.0f);
  outh[idx] = (_Float16)v;
}

// ---------------------------------------------------------------------------
extern "C" void kernel_launch(void* const* d_in, const int* in_sizes, int n_in,
                              void* d_out, int out_size, void* d_ws, size_t ws_size,
                              hipStream_t stream) {
  const float*     x    = (const float*)d_in[0];
  const long long* ei   = (const long long*)d_in[1];   // [2, E] int64
  const float*     text = (const float*)d_in[2];
  const float*     W1   = (const float*)d_in[3];
  const float*     b1   = (const float*)d_in[4];
  const float*     W2   = (const float*)d_in[5];
  const float*     b2   = (const float*)d_in[6];
  const float*     Wc   = (const float*)d_in[7];
  const float*     bc   = (const float*)d_in[8];
  float* out = (float*)d_out;

  const long long* e_src = ei;
  const long long* e_dst = ei + N_EDGES;

  // workspace carving (256B aligned)
  char* w = (char*)d_ws;
  size_t off = 0;
  auto carve = [&](size_t bytes) { void* p = w + off; off = (off + bytes + 255) & ~(size_t)255; return p; };
  float*     deg  = (float*)carve((size_t)N_NODES * 4);
  float*     dis  = (float*)carve((size_t)N_NODES * 4);
  _Float16*  xh   = (_Float16*)carve((size_t)N_NODES * IN_CH * 2);
  _Float16*  w1p  = (_Float16*)carve((size_t)8 * 8  * 512 * 2);   // K=256,N=128
  _Float16*  w2p  = (_Float16*)carve((size_t)4 * 8  * 512 * 2);   // K=128,N=128
  _Float16*  wcp  = (_Float16*)carve((size_t)4 * 64 * 512 * 2);   // K=128,N=1024
  float*     base = (float*)carve((size_t)OUT_PAD * 4);
  float*     bufA = (float*)carve((size_t)N_NODES * HID_CH * 4);  // GEMM out
  float*     bufB = (float*)carve((size_t)N_NODES * HID_CH * 4);  // aggregation
  _Float16*  bufC = (_Float16*)carve((size_t)N_NODES * HID_CH * 2); // f16 act

  const int B = 256;
  // --- normalization ---
  k_fill1    <<<(N_NODES + B - 1) / B, B, 0, stream>>>(deg, N_NODES);
  k_deg_count<<<(N_EDGES + B - 1) / B, B, 0, stream>>>(e_dst, deg);
  k_rsqrt    <<<(N_NODES + B - 1) / B, B, 0, stream>>>(deg, dis, N_NODES);
  // --- f16 / fragment packing ---
  k_f2h<<<(N_NODES * IN_CH + B - 1) / B, B, 0, stream>>>(x, xh, N_NODES * IN_CH);
  k_pack_b<<<(8 * 8  * 512 + B - 1) / B, B, 0, stream>>>(W1, w1p, 8, 8,  HID_CH);
  k_pack_b<<<(4 * 8  * 512 + B - 1) / B, B, 0, stream>>>(W2, w2p, 4, 8,  HID_CH);
  k_pack_b<<<(4 * 64 * 512 + B - 1) / B, B, 0, stream>>>(Wc, wcp, 4, 64, OUT_CH);
  k_base<<<OUT_PAD / B, B, 0, stream>>>(text, Wc, bc, base);

  const int nElem = N_NODES * HID_CH;
  const int eThreads = N_EDGES * 32;
  // --- layer 1: GEMM -> aggregate -> bias+ReLU -> f16 ---
  k_gemm_hid<IN_CH> <<<N_NODES / 16, B, 0, stream>>>(xh, w1p, bufA);
  k_self_init<<<(nElem + B - 1) / B, B, 0, stream>>>(bufA, dis, bufB);
  k_edge_agg <<<(eThreads + B - 1) / B, B, 0, stream>>>(bufA, e_src, e_dst, dis, bufB);
  k_bias_act <<<(nElem + B - 1) / B, B, 0, stream>>>(bufB, b1, bufC, 1);
  // --- layer 2 ---
  k_gemm_hid<HID_CH><<<N_NODES / 16, B, 0, stream>>>(bufC, w2p, bufA);
  k_self_init<<<(nElem + B - 1) / B, B, 0, stream>>>(bufA, dis, bufB);
  k_edge_agg <<<(eThreads + B - 1) / B, B, 0, stream>>>(bufA, e_src, e_dst, dis, bufB);
  k_bias_act <<<(nElem + B - 1) / B, B, 0, stream>>>(bufB, b2, bufC, 0);
  // --- classifier ---
  k_classifier<<<dim3(N_NODES / 16, 2), B, 0, stream>>>(bufC, wcp, base, out);
}
